// Transformer_15934328668792
// MI455X (gfx1250) — compile-verified, matrix operand order checked by reading, not run on hardware
//
#include <hip/hip_runtime.h>
#include <hip/hip_bf16.h>
#include <cstdint>
#include <cstddef>

// ---------------------------------------------------------------------------
// Transformer forward for MI455X (gfx1250): all matmuls via
// v_wmma_f32_16x16x32_f16 (f16 in, f32 accumulate). Weights converted to f16
// transposed once so every WMMA operand streams K-contiguous 16B chunks.
// GEMM uses double-buffered LDS (one barrier per K-step) and, when the
// toolchain exposes it, GLOBAL_LOAD_ASYNC_TO_LDS_B128 for direct HBM->LDS
// staging tracked by ASYNCcnt.
// ---------------------------------------------------------------------------

typedef __attribute__((ext_vector_type(16))) _Float16     v16h;
typedef __attribute__((ext_vector_type(8)))  float        v8f;
typedef __attribute__((ext_vector_type(4)))  unsigned int u32x4;
typedef int i32x4 __attribute__((vector_size(16)));   // matches builtin pointee

constexpr int Bc = 4, Sc = 1024, Dc = 1024, Hc = 16, Lc = 4, Fc = 4096, Vc = 8192;
constexpr int DKc   = Dc / Hc;      // 64
constexpr int Mrows = Bc * Sc;      // 4096
constexpr int BM = 128, BK = 32;
constexpr int LDSS = BK + 8;        // padded LDS stride (80B rows, 16B aligned)

struct Frag16 { union { v16h h; u32x4 u[2]; }; };

// ---- optional gfx1250 async memory->LDS path (compile-guarded) ------------
#if defined(__HIP_DEVICE_COMPILE__) && defined(__has_builtin)
#  if __has_builtin(__builtin_amdgcn_global_load_async_to_lds_b128)
#    define USE_ASYNC_LDS 1
#  endif
#endif
#ifndef USE_ASYNC_LDS
#  define USE_ASYNC_LDS 0
#endif

#if USE_ASYNC_LDS
#  define ASYNC_B128(gsrc, ldst)                                             \
     __builtin_amdgcn_global_load_async_to_lds_b128(                         \
         (__attribute__((address_space(1))) i32x4*)(gsrc),                   \
         (__attribute__((address_space(3))) i32x4*)(ldst), 0, 0)
#  if __has_builtin(__builtin_amdgcn_s_wait_asynccnt)
#    define WAIT_ASYNC() __builtin_amdgcn_s_wait_asynccnt(0)
#  else
#    define WAIT_ASYNC() asm volatile("s_wait_asynccnt 0x0" ::: "memory")
#  endif
#endif

// ---------------------------------------------------------------------------
// Generic batched WMMA GEMM:  C[z] = alpha * A[z] * Bt[z]^T + bias
//   A  : [M,K] f16, row stride lda, batch stride sAz
//   Bt : [N,K] f16 (pre-transposed weights / keys / V^T), row stride ldb
//   C  : optional f32 out, Ch: optional f16 out (same indexing), ldc/sCz
// Block tile 128 x BN_T, 8 waves as 4(M) x 2(N); wave tile 32 x (JT*16)
// -> 2*JT WMMA 16x16x32 per K-step per wave; double-buffered LDS.
// ---------------------------------------------------------------------------
template<int BN_T, int JT>
__global__ __launch_bounds__(256) void wmma_gemm_kernel(
    const _Float16* __restrict__ A,  long long sAz, int lda,
    const _Float16* __restrict__ Bt, long long sBz, int ldb,
    const float* __restrict__ bias,
    float* __restrict__ C, _Float16* __restrict__ Ch,
    long long sCz, int ldc, int K, float alpha, int relu)
{
  static_assert(BN_T == JT * 32, "8 waves: 4xM, 2xN");
  __shared__ __align__(16) _Float16 As[2][BM][LDSS];
  __shared__ __align__(16) _Float16 Bs[2][BN_T][LDSS];

  const int tid  = threadIdx.x;
  const int lane = tid & 31;
  const int wave = tid >> 5;
  const int wm   = wave >> 1;           // 0..3  -> 32 rows each
  const int wn   = wave & 1;            // 0..1  -> JT*16 cols each
  const int z    = blockIdx.z;
  const long long mBase = (long long)blockIdx.x * BM;
  const long long nBase = (long long)blockIdx.y * BN_T;

  A  += (long long)z * sAz;
  Bt += (long long)z * sBz;

  v8f acc[2][JT] = {};

  // staging map: A tile 128x32 halves -> 16 halves/thread
  const int aRow = tid >> 1;            // 0..127
  const int aSeg = (tid & 1) * 16;      // 0 or 16
  // B tile BN_T x 32 halves -> 16 (BN_T=128) or 8 (BN_T=64) halves/thread
  const int bRow = (BN_T == 128) ? (tid >> 1) : (tid >> 2);
  const int bSeg = (BN_T == 128) ? ((tid & 1) * 16) : ((tid & 3) * 8);

  const _Float16* aPtr = A  + (mBase + aRow) * (long long)lda + aSeg;
  const _Float16* bPtr = Bt + (nBase + bRow) * (long long)ldb + bSeg;

  auto stage = [&](int buf) {
#if USE_ASYNC_LDS
    ASYNC_B128(aPtr,     &As[buf][aRow][aSeg]);
    ASYNC_B128(aPtr + 8, &As[buf][aRow][aSeg + 8]);
    if constexpr (BN_T == 128) {
      ASYNC_B128(bPtr,     &Bs[buf][bRow][bSeg]);
      ASYNC_B128(bPtr + 8, &Bs[buf][bRow][bSeg + 8]);
    } else {
      ASYNC_B128(bPtr,     &Bs[buf][bRow][bSeg]);
    }
#else
    u32x4 a0 = *(const u32x4*)(aPtr);
    u32x4 a1 = *(const u32x4*)(aPtr + 8);
    *(u32x4*)(&As[buf][aRow][aSeg])     = a0;
    *(u32x4*)(&As[buf][aRow][aSeg + 8]) = a1;
    if constexpr (BN_T == 128) {
      u32x4 b0 = *(const u32x4*)(bPtr);
      u32x4 b1 = *(const u32x4*)(bPtr + 8);
      *(u32x4*)(&Bs[buf][bRow][bSeg])     = b0;
      *(u32x4*)(&Bs[buf][bRow][bSeg + 8]) = b1;
    } else {
      u32x4 b0 = *(const u32x4*)(bPtr);
      *(u32x4*)(&Bs[buf][bRow][bSeg])     = b0;
    }
#endif
  };

  // fragment map per CDNA5 16-bit A/B layout: lanes 0-15 -> K {0..7,16..23},
  // lanes 16-31 -> K {8..15,24..31}; row/col = lane&15.
  const int koff = (lane >> 4) * 8;     // 0 or 8
  const int lrow = lane & 15;

  stage(0);                              // prologue: tile 0 -> buffer 0
  const int nk = K / BK;

  for (int kt = 0; kt < nk; ++kt) {
    const int p = kt & 1;
#if USE_ASYNC_LDS
    WAIT_ASYNC();                        // this wave's LDS writes landed
#endif
    __syncthreads();                     // single barrier per K-step

    if (kt + 1 < nk) {                   // stage next tile into other buffer
      aPtr += BK; bPtr += BK;
      __builtin_prefetch(aPtr + BK, 0, 1);   // tile after next -> L2
      __builtin_prefetch(bPtr + BK, 0, 1);
      stage(p ^ 1);
    }

    Frag16 af[2], bfr[JT];
    #pragma unroll
    for (int i = 0; i < 2; ++i) {
      const int r = wm * 32 + i * 16 + lrow;
      af[i].u[0] = *(const u32x4*)(&As[p][r][koff]);
      af[i].u[1] = *(const u32x4*)(&As[p][r][16 + koff]);
    }
    #pragma unroll
    for (int j = 0; j < JT; ++j) {
      const int c = wn * (JT * 16) + j * 16 + lrow;
      bfr[j].u[0] = *(const u32x4*)(&Bs[p][c][koff]);
      bfr[j].u[1] = *(const u32x4*)(&Bs[p][c][16 + koff]);
    }
    #pragma unroll
    for (int i = 0; i < 2; ++i)
      #pragma unroll
      for (int j = 0; j < JT; ++j)
        acc[i][j] = __builtin_amdgcn_wmma_f32_16x16x32_f16(
            false, af[i].h, false, bfr[j].h, (short)0, acc[i][j], false, false);
  }

  // epilogue: C/D layout -> lane gives N column, VGPR r gives M row (+8 hi half)
  const long long cz = (long long)z * sCz;
  const int mhalf = (lane >> 4) * 8;
  #pragma unroll
  for (int i = 0; i < 2; ++i) {
    #pragma unroll
    for (int j = 0; j < JT; ++j) {
      const long long n = nBase + wn * (JT * 16) + j * 16 + lrow;
      const float bv = bias ? bias[n] : 0.0f;
      #pragma unroll
      for (int r = 0; r < 8; ++r) {
        const long long m = mBase + wm * 32 + i * 16 + r + mhalf;
        float v = alpha * acc[i][j][r] + bv;
        if (relu) v = fmaxf(v, 0.0f);
        const long long idx = cz + m * (long long)ldc + n;
        if (C)  C[idx]  = v;
        if (Ch) Ch[idx] = (_Float16)v;
      }
    }
  }
}

// ---------------------------------------------------------------------------
// Weight convert + transpose: Wt[n*K + k] = (f16) W[k*N + n]
// ---------------------------------------------------------------------------
__global__ __launch_bounds__(256) void transp_fp16_kernel(
    const float* __restrict__ W, _Float16* __restrict__ Wt, int K, int N)
{
  long long idx = (long long)blockIdx.x * 256 + threadIdx.x;
  if (idx >= (long long)K * N) return;
  int k = (int)(idx / N);
  int n = (int)(idx % N);
  Wt[(long long)n * K + k] = (_Float16)W[idx];
}

// ---------------------------------------------------------------------------
// Embedding + sinusoidal positional encoding -> x (f32) and f16 mirror
// ---------------------------------------------------------------------------
__global__ __launch_bounds__(256) void embed_pe_kernel(
    const int* __restrict__ src, const float* __restrict__ emb,
    float* __restrict__ x, _Float16* __restrict__ x16)
{
  const long long row = blockIdx.x;            // 0..B*S-1
  const int s   = (int)(row % Sc);
  const int tok = src[row];
  const float pos = (float)s;
  #pragma unroll
  for (int i = 0; i < Dc / 256; ++i) {
    const int d = threadIdx.x + i * 256;
    const float arg = (float)(d & ~1) * (-9.210340371976184f / (float)Dc); // -ln(1e4)/D
    const float f = __expf(arg);
    const float pe = (d & 1) ? __cosf(pos * f) : __sinf(pos * f);
    const float v = emb[(long long)tok * Dc + d] + pe;
    x[row * Dc + d]   = v;
    x16[row * Dc + d] = (_Float16)v;
  }
}

// ---------------------------------------------------------------------------
// Masked softmax over one score row [S]; mask = causal(k<=q) & (src[b,q]!=0)
// (reference broadcasts the pad mask over the key axis). Writes f16 probs.
// ---------------------------------------------------------------------------
__global__ __launch_bounds__(256) void softmax_mask_kernel(
    const float* __restrict__ scores, _Float16* __restrict__ probs,
    const int* __restrict__ src, int b, float scale)
{
  const int q = blockIdx.x;
  const int h = blockIdx.y;
  const float* row  = scores + ((long long)h * Sc + q) * Sc;
  _Float16*    prow = probs  + ((long long)h * Sc + q) * Sc;
  const bool qlive = (src[(long long)b * Sc + q] != 0);

  __shared__ float red[256];
  float vals[Sc / 256];
  float mx = -1e30f;
  #pragma unroll
  for (int i = 0; i < Sc / 256; ++i) {
    const int k = threadIdx.x + i * 256;
    const bool ok = qlive && (k <= q);
    const float v = ok ? row[k] * scale : -1e9f;
    vals[i] = v;
    mx = fmaxf(mx, v);
  }
  red[threadIdx.x] = mx; __syncthreads();
  for (int s2 = 128; s2 > 0; s2 >>= 1) {
    if (threadIdx.x < s2) red[threadIdx.x] = fmaxf(red[threadIdx.x], red[threadIdx.x + s2]);
    __syncthreads();
  }
  mx = red[0]; __syncthreads();

  float sum = 0.0f;
  #pragma unroll
  for (int i = 0; i < Sc / 256; ++i) { vals[i] = __expf(vals[i] - mx); sum += vals[i]; }
  red[threadIdx.x] = sum; __syncthreads();
  for (int s2 = 128; s2 > 0; s2 >>= 1) {
    if (threadIdx.x < s2) red[threadIdx.x] += red[threadIdx.x + s2];
    __syncthreads();
  }
  const float inv = 1.0f / red[0];
  #pragma unroll
  for (int i = 0; i < Sc / 256; ++i)
    prow[threadIdx.x + i * 256] = (_Float16)(vals[i] * inv);
}

// ---------------------------------------------------------------------------
// V transpose per head: vT[b][h][dk][s] = v[b][s][h*DK+dk]
// ---------------------------------------------------------------------------
__global__ __launch_bounds__(256) void transpose_v_kernel(
    const _Float16* __restrict__ v, _Float16* __restrict__ vT)
{
  long long idx = (long long)blockIdx.x * 256 + threadIdx.x;   // B*S*D elems
  const int d = (int)(idx % Dc);
  const long long bs = idx / Dc;
  const int s = (int)(bs % Sc);
  const int b = (int)(bs / Sc);
  const int h = d / DKc, dk = d % DKc;
  vT[(((long long)b * Hc + h) * DKc + dk) * Sc + s] = v[idx];
}

// ---------------------------------------------------------------------------
// Residual add + LayerNorm; writes f32 and f16 mirror (may alias x input)
// ---------------------------------------------------------------------------
__global__ __launch_bounds__(256) void add_ln_kernel(
    const float* __restrict__ x, const float* __restrict__ y,
    const float* __restrict__ g, const float* __restrict__ beta,
    float* __restrict__ xo, _Float16* __restrict__ xo16)
{
  const long long rowb = (long long)blockIdx.x * Dc;
  __shared__ float red[256];
  float v[Dc / 256];
  float s = 0.0f;
  #pragma unroll
  for (int i = 0; i < Dc / 256; ++i) {
    const int d = threadIdx.x + i * 256;
    v[i] = x[rowb + d] + y[rowb + d];
    s += v[i];
  }
  red[threadIdx.x] = s; __syncthreads();
  for (int s2 = 128; s2 > 0; s2 >>= 1) {
    if (threadIdx.x < s2) red[threadIdx.x] += red[threadIdx.x + s2];
    __syncthreads();
  }
  const float mean = red[0] * (1.0f / Dc); __syncthreads();

  float s2sum = 0.0f;
  #pragma unroll
  for (int i = 0; i < Dc / 256; ++i) { const float dd = v[i] - mean; s2sum += dd * dd; }
  red[threadIdx.x] = s2sum; __syncthreads();
  for (int s2 = 128; s2 > 0; s2 >>= 1) {
    if (threadIdx.x < s2) red[threadIdx.x] += red[threadIdx.x + s2];
    __syncthreads();
  }
  const float inv = rsqrtf(red[0] * (1.0f / Dc) + 1e-5f);

  #pragma unroll
  for (int i = 0; i < Dc / 256; ++i) {
    const int d = threadIdx.x + i * 256;
    const float o = g[d] * (v[i] - mean) * inv + beta[d];
    xo[rowb + d]   = o;
    xo16[rowb + d] = (_Float16)o;
  }
}

// ---------------------------------------------------------------------------
// Host orchestration
// ---------------------------------------------------------------------------
extern "C" void kernel_launch(void* const* d_in, const int* in_sizes, int n_in,
                              void* d_out, int out_size, void* d_ws, size_t ws_size,
                              hipStream_t stream) {
  (void)in_sizes; (void)n_in; (void)out_size; (void)ws_size;

  const int*   src   = (const int*)  d_in[0];
  const float* emb   = (const float*)d_in[1];
  const float* Wq    = (const float*)d_in[2];
  const float* bq    = (const float*)d_in[3];
  const float* Wk    = (const float*)d_in[4];
  const float* bk    = (const float*)d_in[5];
  const float* Wv    = (const float*)d_in[6];
  const float* bv    = (const float*)d_in[7];
  const float* Wo    = (const float*)d_in[8];
  const float* bo    = (const float*)d_in[9];
  const float* W1    = (const float*)d_in[10];
  const float* b1    = (const float*)d_in[11];
  const float* W2    = (const float*)d_in[12];
  const float* b2    = (const float*)d_in[13];
  const float* g1    = (const float*)d_in[14];
  const float* beta1 = (const float*)d_in[15];
  const float* g3    = (const float*)d_in[16];
  const float* beta3 = (const float*)d_in[17];
  const float* Wf    = (const float*)d_in[18];
  const float* bf    = (const float*)d_in[19];

  // workspace carving (256B aligned)
  size_t off = 0;
  auto wsAlloc = [&](size_t bytes) -> void* {
    void* p = (char*)d_ws + off;
    off += (bytes + 255) & ~(size_t)255;
    return p;
  };
  float*    xf32   = (float*)   wsAlloc(sizeof(float) * (size_t)Mrows * Dc);
  _Float16* xf16   = (_Float16*)wsAlloc(2ull * Mrows * Dc);
  _Float16* qh     = (_Float16*)wsAlloc(2ull * Mrows * Dc);
  _Float16* kh     = (_Float16*)wsAlloc(2ull * Mrows * Dc);
  _Float16* vh     = (_Float16*)wsAlloc(2ull * Mrows * Dc);
  _Float16* vTh    = (_Float16*)wsAlloc(2ull * Mrows * Dc);      // [B,H,DK,S]
  _Float16* attnH  = (_Float16*)wsAlloc(2ull * Mrows * Dc);
  float*    obuf   = (float*)   wsAlloc(sizeof(float) * (size_t)Mrows * Dc);
  float*    ffbuf  = (float*)   wsAlloc(sizeof(float) * (size_t)Mrows * Dc);
  _Float16* h1H    = (_Float16*)wsAlloc(2ull * Mrows * Fc);
  float*    scoresF= (float*)   wsAlloc(sizeof(float) * (size_t)Hc * Sc * Sc); // per-b reuse
  _Float16* probsH = (_Float16*)wsAlloc(2ull * Hc * Sc * Sc);                  // per-b reuse
  _Float16* WqT    = (_Float16*)wsAlloc(2ull * Lc * Dc * Dc);
  _Float16* WkT    = (_Float16*)wsAlloc(2ull * Lc * Dc * Dc);
  _Float16* WvT    = (_Float16*)wsAlloc(2ull * Lc * Dc * Dc);
  _Float16* WoT    = (_Float16*)wsAlloc(2ull * Lc * Dc * Dc);
  _Float16* W1T    = (_Float16*)wsAlloc(2ull * Lc * Fc * Dc);    // [F,D] per layer
  _Float16* W2T    = (_Float16*)wsAlloc(2ull * Lc * Dc * Fc);    // [D,F] per layer
  _Float16* WfT    = (_Float16*)wsAlloc(2ull * Vc * Dc);         // [V,D]

  auto transp = [&](const float* W, _Float16* Wt, int K, int N) {
    long long tot = (long long)K * N;
    transp_fp16_kernel<<<(unsigned)((tot + 255) / 256), 256, 0, stream>>>(W, Wt, K, N);
  };
  auto gemm = [&](const _Float16* A, long long sAz, int lda,
                  const _Float16* Bt, long long sBz, int ldb,
                  const float* bias, float* C, _Float16* Ch,
                  long long sCz, int ldc,
                  int M, int N, int K, int nz, float alpha, int relu) {
    if (N % 128 == 0) {
      dim3 grid((unsigned)(M / BM), (unsigned)(N / 128), (unsigned)nz);
      wmma_gemm_kernel<128, 4><<<grid, 256, 0, stream>>>(
          A, sAz, lda, Bt, sBz, ldb, bias, C, Ch, sCz, ldc, K, alpha, relu);
    } else {
      dim3 grid((unsigned)(M / BM), (unsigned)(N / 64), (unsigned)nz);
      wmma_gemm_kernel<64, 2><<<grid, 256, 0, stream>>>(
          A, sAz, lda, Bt, sBz, ldb, bias, C, Ch, sCz, ldc, K, alpha, relu);
    }
  };

  // ---- one-time weight conversion (f32 -> f16, transposed to [N,K]) ----
  for (int l = 0; l < Lc; ++l) {
    transp(Wq + (size_t)l * Dc * Dc, WqT + (size_t)l * Dc * Dc, Dc, Dc);
    transp(Wk + (size_t)l * Dc * Dc, WkT + (size_t)l * Dc * Dc, Dc, Dc);
    transp(Wv + (size_t)l * Dc * Dc, WvT + (size_t)l * Dc * Dc, Dc, Dc);
    transp(Wo + (size_t)l * Dc * Dc, WoT + (size_t)l * Dc * Dc, Dc, Dc);
    transp(W1 + (size_t)l * Dc * Fc, W1T + (size_t)l * Fc * Dc, Dc, Fc);
    transp(W2 + (size_t)l * Fc * Dc, W2T + (size_t)l * Dc * Fc, Fc, Dc);
  }
  transp(Wf, WfT, Dc, Vc);

  // ---- embedding + positional encoding ----
  embed_pe_kernel<<<Mrows, 256, 0, stream>>>(src, emb, xf32, xf16);

  const float scale = 0.125f;  // 1/sqrt(64)

  for (int l = 0; l < Lc; ++l) {
    const float* bq_l = bq + (size_t)l * Dc;
    const float* bk_l = bk + (size_t)l * Dc;
    const float* bv_l = bv + (size_t)l * Dc;
    const float* bo_l = bo + (size_t)l * Dc;
    const float* b1_l = b1 + (size_t)l * Fc;
    const float* b2_l = b2 + (size_t)l * Dc;

    // Q, K, V projections (f16 outputs)
    gemm(xf16, 0, Dc, WqT + (size_t)l * Dc * Dc, 0, Dc, bq_l,
         nullptr, qh, 0, Dc, Mrows, Dc, Dc, 1, 1.0f, 0);
    gemm(xf16, 0, Dc, WkT + (size_t)l * Dc * Dc, 0, Dc, bk_l,
         nullptr, kh, 0, Dc, Mrows, Dc, Dc, 1, 1.0f, 0);
    gemm(xf16, 0, Dc, WvT + (size_t)l * Dc * Dc, 0, Dc, bv_l,
         nullptr, vh, 0, Dc, Mrows, Dc, Dc, 1, 1.0f, 0);

    transpose_v_kernel<<<(Mrows * Dc) / 256, 256, 0, stream>>>(vh, vTh);

    for (int b = 0; b < Bc; ++b) {
      // scores[h] = scale * Q_bh (SxDK) * K_bh^T   -> f32 [H,S,S]
      gemm(qh + (size_t)b * Sc * Dc, DKc, Dc,
           kh + (size_t)b * Sc * Dc, DKc, Dc,
           nullptr, scoresF, nullptr, (long long)Sc * Sc, Sc,
           Sc, Sc, DKc, Hc, scale, 0);

      softmax_mask_kernel<<<dim3(Sc, Hc), 256, 0, stream>>>(scoresF, probsH, src, b, 1.0f);

      // attn[h] = P_bh (SxS) * V_bh (SxDK)  -> f16 into attnH[b, s, h*DK+dk]
      gemm(probsH, (long long)Sc * Sc, Sc,
           vTh + (size_t)b * Hc * DKc * Sc, (long long)DKc * Sc, Sc,
           nullptr, nullptr, attnH + (size_t)b * Sc * Dc, DKc, Dc,
           Sc, DKc, Sc, Hc, 1.0f, 0);
    }

    // output projection + residual LN
    gemm(attnH, 0, Dc, WoT + (size_t)l * Dc * Dc, 0, Dc, bo_l,
         obuf, nullptr, 0, Dc, Mrows, Dc, Dc, 1, 1.0f, 0);
    add_ln_kernel<<<Mrows, 256, 0, stream>>>(xf32, obuf,
        g1 + (size_t)l * Dc, beta1 + (size_t)l * Dc, xf32, xf16);

    // FFN: relu(x W1 + b1) W2 + b2, residual LN
    gemm(xf16, 0, Dc, W1T + (size_t)l * Fc * Dc, 0, Dc, b1_l,
         nullptr, h1H, 0, Fc, Mrows, Fc, Dc, 1, 1.0f, 1);
    gemm(h1H, 0, Fc, W2T + (size_t)l * Dc * Fc, 0, Fc, b2_l,
         ffbuf, nullptr, 0, Dc, Mrows, Dc, Fc, 1, 1.0f, 0);
    add_ln_kernel<<<Mrows, 256, 0, stream>>>(xf32, ffbuf,
        g3 + (size_t)l * Dc, beta3 + (size_t)l * Dc, xf32, xf16);
  }

  // final vocab projection -> f32 logits in d_out
  gemm(xf16, 0, Dc, WfT, 0, Dc, bf,
       (float*)d_out, nullptr, 0, Vc, Mrows, Vc, Dc, 1, 1.0f, 0);
}
